// MultiHeadAttention_7919919694506
// MI455X (gfx1250) — matrix operand order repaired and checked by reading, not compiled, not run on hardware
//
#include <hip/hip_runtime.h>

// ---------------------------------------------------------------------------
// MHA forward for MI455X (gfx1250): bf16 WMMA GEMMs + flash-style attention.
// Tile staging uses CDNA5 async global->LDS loads (ASYNCcnt) via inline asm.
// ---------------------------------------------------------------------------

#define T_SEQ 4096
#define C_DIM 1024
#define N_HEAD 16
#define HEAD_DIM 64

typedef __attribute__((ext_vector_type(16))) __bf16 v16bf;
typedef __attribute__((ext_vector_type(8)))  float  v8f;

union FragB16 { v16bf v; unsigned int u[8]; unsigned short h[16]; };
union AccF    { v8f   v; float f[8]; };

__device__ __forceinline__ unsigned short f2bf(float x) {
  union { float f; unsigned int u; } c; c.f = x;
  unsigned int r = c.u + 0x7FFFu + ((c.u >> 16) & 1u); // round-to-nearest-even
  return (unsigned short)(r >> 16);
}

// CDNA5 async global->LDS copy (16B per lane), tracked with ASYNCcnt.
__device__ __forceinline__ void async_ld_b128(unsigned lds_off, const void* g) {
  asm volatile("global_load_async_to_lds_b128 %0, %1, off"
               :: "v"(lds_off), "v"(g) : "memory");
}
__device__ __forceinline__ void wait_async0() {
  asm volatile("s_wait_asynccnt 0" ::: "memory");
}
__device__ __forceinline__ unsigned lds_off(const void* p) {
  return (unsigned)(unsigned long long)p;  // LDS aperture: offset in addr[31:0]
}

// A-fragment K index for 16-bit 16x32 A tiles (ISA 7.12.2):
//   vgpr p, lanes 0-15: K = {0,1},{2,3},{4,5},{6,7},{16,17},...  lanes 16-31: +8
__device__ __forceinline__ int ka_idx(int p, int hi) {
  return 2 * (p & 3) + ((p >> 2) << 4) + (hi << 3);
}
// B-fragment K index for 16-bit 32x16 B tiles (column per lane):
//   vgpr p, lanes 0-15: K = {0,1},{2,3},{4,5},{6,7},{16,17},...  lanes 16-31: +8
__device__ __forceinline__ int kb_idx(int p, int hi) {
  return 2 * (p & 3) + ((p >> 2) << 4) + (hi << 3);
}

// ---------------------------------------------------------------------------
// fp32 -> bf16 conversion
// ---------------------------------------------------------------------------
__global__ __launch_bounds__(256) void cvt_f32_bf16(
    const float* __restrict__ src, unsigned short* __restrict__ dst, int n) {
  int i = blockIdx.x * 256 + threadIdx.x;
  if (i < n) dst[i] = f2bf(src[i]);
}

// ---------------------------------------------------------------------------
// GEMM: out[M,N] = A[M,K] * W[N,K]^T + bias[N]
// Block: 256 threads (8 waves, 4x2). Tile: 128(M) x 128(N). K stepped by 32.
// Each wave owns a 32x64 strip -> 8 WMMA per K-step (2 A-frags x 4 B-frags).
// ---------------------------------------------------------------------------
template <bool OUT_BF16>
__global__ __launch_bounds__(256) void gemm_wmma(
    const unsigned short* __restrict__ A,  // [M,K] bf16
    const unsigned short* __restrict__ W,  // [N,K] bf16
    const float* __restrict__ bias,        // [N]
    void* __restrict__ out, int M, int N, int K) {
  constexpr int LS = 40;  // padded LDS stride (halves): conflict-free, 16B rows
  __shared__ unsigned short sA[128 * LS];
  __shared__ unsigned short sB[128 * LS];

  const int tid  = threadIdx.x;
  const int wave = tid >> 5;
  const int lane = tid & 31;
  const int lh   = lane & 15;
  const int hi   = (lane >> 4) & 1;
  const int wm   = wave & 3;   // 4 waves across M (32 rows each)
  const int wn   = wave >> 2;  // 2 waves across N (64 cols each)
  const int n0   = blockIdx.x * 128;
  const int m0   = blockIdx.y * 128;

  AccF acc[2][4];
#pragma unroll
  for (int i = 0; i < 2; ++i)
#pragma unroll
    for (int j = 0; j < 4; ++j)
#pragma unroll
      for (int r = 0; r < 8; ++r) acc[i][j].f[r] = 0.0f;

  for (int kk = 0; kk < K; kk += 32) {
    // Async-stage A and B tiles (128x32 each): 512 x b128 per tile, 2+2/thread
#pragma unroll
    for (int i = 0; i < 2; ++i) {
      int idx = tid + i * 256;
      int row = idx >> 2, seg = idx & 3;
      async_ld_b128(lds_off(sA + row * LS + seg * 8),
                    A + (size_t)(m0 + row) * K + kk + seg * 8);
      async_ld_b128(lds_off(sB + row * LS + seg * 8),
                    W + (size_t)(n0 + row) * K + kk + seg * 8);
    }
    wait_async0();
    __syncthreads();

    FragB16 a[2];
#pragma unroll
    for (int i = 0; i < 2; ++i) {
      const int arow = wm * 32 + i * 16 + lh;
#pragma unroll
      for (int p = 0; p < 8; ++p)
        a[i].u[p] = *(const unsigned int*)(sA + arow * LS + ka_idx(p, hi));
    }
#pragma unroll
    for (int j = 0; j < 4; ++j) {
      FragB16 b;
      const int brow = wn * 64 + j * 16 + lh;
#pragma unroll
      for (int p = 0; p < 8; ++p)
        b.u[p] = *(const unsigned int*)(sB + brow * LS + kb_idx(p, hi));
      acc[0][j].v = __builtin_amdgcn_wmma_f32_16x16x32_bf16(
          false, a[0].v, false, b.v, (short)0, acc[0][j].v, false, false);
      acc[1][j].v = __builtin_amdgcn_wmma_f32_16x16x32_bf16(
          false, a[1].v, false, b.v, (short)0, acc[1][j].v, false, false);
    }
    __syncthreads();
  }

  // Store: C/D layout -> lane holds (M = r + 8*hi, N = lh) per vgpr r
#pragma unroll
  for (int j = 0; j < 4; ++j) {
    const int col = n0 + wn * 64 + j * 16 + lh;
    const float bv = bias[col];
#pragma unroll
    for (int i = 0; i < 2; ++i)
#pragma unroll
      for (int r = 0; r < 8; ++r) {
        const int row = m0 + wm * 32 + i * 16 + r + hi * 8;
        const float val = acc[i][j].f[r] + bv;
        if (OUT_BF16)
          ((unsigned short*)out)[(size_t)row * N + col] = f2bf(val);
        else
          ((float*)out)[(size_t)row * N + col] = val;
      }
  }
}

// ---------------------------------------------------------------------------
// Causal flash attention. Block = 1 head x 64 query rows (4 waves x 16 rows).
// Q/K/V/Y are bf16 [T, C] with head h occupying columns [64h, 64h+64).
// ---------------------------------------------------------------------------
__global__ __launch_bounds__(128) void attn_wmma(
    const unsigned short* __restrict__ Q, const unsigned short* __restrict__ K,
    const unsigned short* __restrict__ V, unsigned short* __restrict__ Y) {
  constexpr int LK = 72;  // padded stride (halves): conflict-free, 16B rows
  __shared__ unsigned short sK[32 * LK];
  __shared__ unsigned short sV[32 * LK];
  __shared__ unsigned short sP[4][16 * 40];  // per-wave P scratch (16x32)

  const int tid  = threadIdx.x;
  const int wave = tid >> 5;
  const int lane = tid & 31;
  const int lh   = lane & 15;
  const int hi   = (lane >> 4) & 1;
  const int head = blockIdx.y;
  const int qb   = blockIdx.x;           // 64-row query block
  const int q0   = qb * 64 + wave * 16;  // this wave's first query row
  const int c0   = head * HEAD_DIM;

  // Q fragments: one 16x64 A strip = two 16x32 A fragments (kept in VGPRs)
  FragB16 qa[2];
  {
    const int row = q0 + lh;
#pragma unroll
    for (int h2 = 0; h2 < 2; ++h2)
#pragma unroll
      for (int p = 0; p < 8; ++p) {
        int k = ka_idx(p, hi) + h2 * 32;
        qa[h2].u[p] = *(const unsigned int*)(Q + (size_t)row * C_DIM + c0 + k);
      }
  }

  AccF o[4];
#pragma unroll
  for (int j = 0; j < 4; ++j)
#pragma unroll
    for (int r = 0; r < 8; ++r) o[j].f[r] = 0.0f;
  float mrow[8], lrow[8];
#pragma unroll
  for (int r = 0; r < 8; ++r) { mrow[r] = -1e30f; lrow[r] = 0.0f; }

  const float scale = 0.125f;        // 1/sqrt(64)
  const int   nkb   = qb * 2 + 2;    // causal: key blocks of 32 covering q rows

  for (int kb = 0; kb < nkb; ++kb) {
    const int kbase = kb * 32;
    // Async-stage K,V blocks (32x64 each): 256 x b128 per matrix, 2+2/thread
#pragma unroll
    for (int i = 0; i < 2; ++i) {
      int idx = tid + i * 128;
      int row = idx >> 3, seg = idx & 7;
      async_ld_b128(lds_off(sK + row * LK + seg * 8),
                    K + (size_t)(kbase + row) * C_DIM + c0 + seg * 8);
      async_ld_b128(lds_off(sV + row * LK + seg * 8),
                    V + (size_t)(kbase + row) * C_DIM + c0 + seg * 8);
    }
    wait_async0();
    __syncthreads();

    // S = Q * K^T : two 16(q) x 16(k) tiles, each accumulated over d=64
    AccF s[2];
#pragma unroll
    for (int t = 0; t < 2; ++t) {
#pragma unroll
      for (int r = 0; r < 8; ++r) s[t].f[r] = 0.0f;
#pragma unroll
      for (int h2 = 0; h2 < 2; ++h2) {
        FragB16 b;  // column n = key row (t*16 + lh); K dim = head dim
        const int krow = t * 16 + lh;
#pragma unroll
        for (int p = 0; p < 8; ++p) {
          int k = kb_idx(p, hi) + h2 * 32;
          b.u[p] = *(const unsigned int*)(sK + krow * LK + k);
        }
        s[t].v = __builtin_amdgcn_wmma_f32_16x16x32_bf16(
            false, qa[h2].v, false, b.v, (short)0, s[t].v, false, false);
      }
    }

    // Causal mask + online softmax update (rows live in vgpr r, cols in lanes)
#pragma unroll
    for (int r = 0; r < 8; ++r) {
      const int qrow = q0 + r + hi * 8;
      float v0 = s[0].f[r] * scale;
      float v1 = s[1].f[r] * scale;
      if (kbase + lh > qrow)      v0 = -1e30f;
      if (kbase + 16 + lh > qrow) v1 = -1e30f;
      float mx = fmaxf(v0, v1);
#pragma unroll
      for (int d = 8; d >= 1; d >>= 1) mx = fmaxf(mx, __shfl_xor(mx, d, 32));
      const float mnew  = fmaxf(mrow[r], mx);
      const float alpha = __expf(mrow[r] - mnew);
      const float p0 = __expf(v0 - mnew);
      const float p1 = __expf(v1 - mnew);
      s[0].f[r] = p0;
      s[1].f[r] = p1;
      float rs = p0 + p1;
#pragma unroll
      for (int d = 8; d >= 1; d >>= 1) rs += __shfl_xor(rs, d, 32);
      lrow[r] = lrow[r] * alpha + rs;
      mrow[r] = mnew;
#pragma unroll
      for (int j = 0; j < 4; ++j) o[j].f[r] *= alpha;
    }

    // P (16x32, C/D layout) -> LDS -> reload as A fragment
    unsigned short* Pw = sP[wave];
#pragma unroll
    for (int t = 0; t < 2; ++t)
#pragma unroll
      for (int r = 0; r < 8; ++r)
        Pw[(r + hi * 8) * 40 + t * 16 + lh] = f2bf(s[t].f[r]);
    __syncthreads();

    FragB16 pa;
#pragma unroll
    for (int p = 0; p < 8; ++p)
      pa.u[p] = *(const unsigned int*)(Pw + lh * 40 + ka_idx(p, hi));

    // O += P * V : 4 output column tiles of 16
#pragma unroll
    for (int j = 0; j < 4; ++j) {
      FragB16 b;
      const int col = j * 16 + lh;
#pragma unroll
      for (int p = 0; p < 8; ++p) {
        const int k = kb_idx(p, hi);
        unsigned int e0 = sV[(size_t)k * LK + col];
        unsigned int e1 = sV[(size_t)(k + 1) * LK + col];
        b.u[p] = e0 | (e1 << 16);
      }
      o[j].v = __builtin_amdgcn_wmma_f32_16x16x32_bf16(
          false, pa.v, false, b.v, (short)0, o[j].v, false, false);
    }
    __syncthreads();
  }

  // Epilogue: normalize and store Y (bf16)
#pragma unroll
  for (int j = 0; j < 4; ++j) {
    const int col = c0 + j * 16 + lh;
#pragma unroll
    for (int r = 0; r < 8; ++r) {
      const int row = q0 + r + hi * 8;
      Y[(size_t)row * C_DIM + col] = f2bf(o[j].f[r] / lrow[r]);
    }
  }
}

// ---------------------------------------------------------------------------
// Launch
// ---------------------------------------------------------------------------
extern "C" void kernel_launch(void* const* d_in, const int* in_sizes, int n_in,
                              void* d_out, int out_size, void* d_ws,
                              size_t ws_size, hipStream_t stream) {
  const float* x    = (const float*)d_in[0];
  const float* wq_w = (const float*)d_in[1];
  const float* wq_b = (const float*)d_in[2];
  const float* wk_w = (const float*)d_in[3];
  const float* wk_b = (const float*)d_in[4];
  const float* wv_w = (const float*)d_in[5];
  const float* wv_b = (const float*)d_in[6];
  const float* wo_w = (const float*)d_in[7];
  const float* wo_b = (const float*)d_in[8];

  const size_t TC = (size_t)T_SEQ * C_DIM;  // 4M
  const size_t CC = (size_t)C_DIM * C_DIM;  // 1M

  unsigned short* xb  = (unsigned short*)d_ws;
  unsigned short* wqb = xb + TC;
  unsigned short* wkb = wqb + CC;
  unsigned short* wvb = wkb + CC;
  unsigned short* wob = wvb + CC;
  unsigned short* Qb  = wob + CC;
  unsigned short* Kb  = Qb + TC;
  unsigned short* Vb  = Kb + TC;
  unsigned short* Yb  = Vb + TC;

  // fp32 -> bf16 conversions
  cvt_f32_bf16<<<(int)((TC + 255) / 256), 256, 0, stream>>>(x, xb, (int)TC);
  cvt_f32_bf16<<<(int)((CC + 255) / 256), 256, 0, stream>>>(wq_w, wqb, (int)CC);
  cvt_f32_bf16<<<(int)((CC + 255) / 256), 256, 0, stream>>>(wk_w, wkb, (int)CC);
  cvt_f32_bf16<<<(int)((CC + 255) / 256), 256, 0, stream>>>(wv_w, wvb, (int)CC);
  cvt_f32_bf16<<<(int)((CC + 255) / 256), 256, 0, stream>>>(wo_w, wob, (int)CC);

  // Q/K/V projections: [T,C] = x @ W^T + b  (bf16 output)
  dim3 ggrid(C_DIM / 128, T_SEQ / 128);
  gemm_wmma<true><<<ggrid, 256, 0, stream>>>(xb, wqb, wq_b, Qb, T_SEQ, C_DIM, C_DIM);
  gemm_wmma<true><<<ggrid, 256, 0, stream>>>(xb, wkb, wk_b, Kb, T_SEQ, C_DIM, C_DIM);
  gemm_wmma<true><<<ggrid, 256, 0, stream>>>(xb, wvb, wv_b, Vb, T_SEQ, C_DIM, C_DIM);

  // Causal attention per head
  attn_wmma<<<dim3(T_SEQ / 64, N_HEAD), 128, 0, stream>>>(Qb, Kb, Vb, Yb);

  // Output projection (fp32 output straight to d_out)
  gemm_wmma<false><<<ggrid, 256, 0, stream>>>(Yb, wob, wo_b, d_out, T_SEQ, C_DIM, C_DIM);
}
